// VectorQuantisedVAE_88295937671316
// MI455X (gfx1250) — matrix operand order repaired and checked by reading, not compiled, not run on hardware
//
#include <hip/hip_runtime.h>
#include <hip/hip_bf16.h>

typedef unsigned short u16;
typedef __attribute__((ext_vector_type(16))) __bf16        v16bf;
typedef __attribute__((ext_vector_type(8)))  float         v8f;
typedef __attribute__((ext_vector_type(4)))  unsigned int  u32x4;
typedef __attribute__((ext_vector_type(4)))  int           i32x4;
typedef __attribute__((ext_vector_type(8)))  int           i32x8;

union Frag {
    u32x4 q[2];   // 32 bytes
    v16bf bf;     // 16 x bf16
};

#if defined(__has_builtin)
#if __has_builtin(__builtin_amdgcn_tensor_load_to_lds) && \
    __has_builtin(__builtin_amdgcn_s_wait_tensorcnt)
#define HAVE_TDM 1
#endif
#endif

__device__ __forceinline__ u16 f2bf(float f) {
    unsigned int u = __float_as_uint(f);
    unsigned int r = u + 0x7FFFu + ((u >> 16) & 1u);   // round-to-nearest-even
    return (u16)(r >> 16);
}

// ---------------------------------------------------------------------------
// fp32 -> bf16 convert
// ---------------------------------------------------------------------------
__global__ void f32_to_bf16_kernel(const float* __restrict__ in, u16* __restrict__ out, int n) {
    int i = blockIdx.x * blockDim.x + threadIdx.x;
    if (i < n) out[i] = f2bf(in[i]);
}

// ---------------------------------------------------------------------------
// Pack weights fp32 [OC][IC][KH][KW] -> bf16 [OC][K], K=(ic*KH+kh)*KW+kw,
// optional spatial flip (transposed conv == lhs-dilated conv w/ flipped k).
// Row-major [OC][K]: rows contiguous in K -> TDM/vector-friendly tiles.
// ---------------------------------------------------------------------------
__global__ void pack_w_kernel(const float* __restrict__ w, u16* __restrict__ wp,
                              int IC, int KH, int KW, int flip, int n) {
    int i = blockIdx.x * blockDim.x + threadIdx.x;
    if (i >= n) return;
    int K  = IC * KH * KW;
    int oc = i / K, k = i - oc * K;
    int kk = KH * KW;
    int ic = k / kk; int r = k - ic * kk;
    int kh = r / KW; int kw = r - kh * KW;
    if (flip) { kh = KH - 1 - kh; kw = KW - 1 - kw; }
    wp[i] = f2bf(w[((oc * IC + ic) * KH + kh) * KW + kw]);
}

// ---------------------------------------------------------------------------
// Implicit-GEMM conv via bf16 WMMA, f32 accumulate. Roles:
//   A (16x32 frag) = weights Wbf[OC][K]   (row-major, contiguous K)
//   B (32x16 frag) = im2col pixels Ps[pix][K]
//   D[oc][pix] scattered to NCHW with shift/mask math (all dims pow2).
// Block: 256 threads (8 waves). Tile 64(OC) x 128(pix), BK=64.
// Wave grid 2x4; each wave owns 32x32 -> 4 accumulators, 8 WMMAs per stage.
// Double-buffered software pipeline: stage kb+1 (TDM weight DMA + pixel
// gather) overlaps the 8 WMMAs of stage kb; one barrier per stage.
// ---------------------------------------------------------------------------
__global__ __launch_bounds__(256) void conv_wmma_kernel(
    const u16* __restrict__ Xb, const u16* __restrict__ Wbf,
    const float* __restrict__ bias,
    float* __restrict__ outF, u16* __restrict__ outB,
    int Bn, int IC, int IH, int IW,
    int OC, int owSh, int ohowSh,
    int kwSh, int khkwSh,
    int stride, int pad, int dil, int act)
{
    __shared__ alignas(16) u16 Ws[2][64][72];   // [buf][oc][K=64]+8 pad, row 144B
    __shared__ alignas(16) u16 Ps[2][128][72];  // [buf][pix][K=64]+8 pad

    const int owMask   = (1 << owSh) - 1;
    const int ohowMask = (1 << ohowSh) - 1;
    const int kwMask   = (1 << kwSh) - 1;
    const int khkwMask = (1 << khkwSh) - 1;

    const int Mpix = Bn << ohowSh;
    const int K    = IC << khkwSh;
    const int oc0  = blockIdx.x << 6;   // *64  (OC tile)
    const int p0   = blockIdx.y << 7;   // *128 (pixel tile)

    const int tid   = threadIdx.x;
    const int lane  = tid & 31;
    const int wv    = tid >> 5;
    const int wmoc  = (wv & 1) << 5;    // 0,32   (oc sub-tile)
    const int wnpix = (wv >> 1) << 5;   // 0..96  (pixel sub-tile)
    const int lhalf = lane >> 4;        // 0 or 1
    const int lmod  = lane & 15;

    v8f acc00 = {}, acc01 = {}, acc10 = {}, acc11 = {};

    // stage-invariant per-thread coordinates
    const int kcP = tid & 63;           // pixel-stage K column
    const int prP = tid >> 6;           // pixel-stage row base (step 4)
    const int wrW = tid >> 3;           // weight-stage oc row (fallback, 0..31)
    const int wcW = (tid & 7) << 3;     // weight-stage K chunk base (halves)

#if defined(HAVE_TDM)
    // Static D# group1: data_size=1 (2B), pad_enable=1,
    // pad_interval=4 (32 dwords = 64 halves), pad_amount=3 (4 dwords = 8 halves).
    i32x8 g1s = {};
    g1s[0] = (1 << 16) | (1 << 20) | (4 << 22) | (3 << 25);
    g1s[1] = (int)(((unsigned)K & 0xFFFFu) << 16);                          // tensor_dim0 lo
    g1s[2] = (int)(((unsigned)K >> 16) | (((unsigned)OC & 0xFFFFu) << 16)); // dim0 hi | dim1 lo
    g1s[3] = (int)(((unsigned)OC >> 16) | (64u << 16));                     // dim1 hi | tile_dim0=64
    g1s[4] = 64;                                                            // tile_dim1=64, tile_dim2=0
    g1s[5] = (int)(unsigned)K;                                              // tensor_dim0_stride lo
    g1s[6] = 0; g1s[7] = 0;
    const unsigned ldsWs0 = (unsigned)(unsigned long long)(const void*)&Ws[0][0][0];
    const unsigned ldsWs1 = (unsigned)(unsigned long long)(const void*)&Ws[1][0][0];
    const i32x4 gz4 = {};
    const i32x8 gz8 = {};
#endif

    // ---- staging helpers ----
    auto stageW = [&](int k0, int bufIdx) {
#if defined(HAVE_TDM)
        if (wv == 0) {
            unsigned long long ga =
                (unsigned long long)(const void*)(Wbf + (size_t)oc0 * K + k0);
            u32x4 g0;
            g0.x = 1u;                                   // count=1 valid descriptor
            g0.y = bufIdx ? ldsWs1 : ldsWs0;             // LDS byte address
            g0.z = (unsigned)(ga & 0xFFFFFFFFu);         // global_addr[31:0]
            g0.w = (unsigned)((ga >> 32) & 0x01FFFFFFu) | (2u << 30);  // hi | type=2
#if __clang_major__ >= 23
            __builtin_amdgcn_tensor_load_to_lds(g0, g1s, gz4, gz4, gz8, 0);
#else
            __builtin_amdgcn_tensor_load_to_lds(g0, g1s, gz4, gz4, 0);
#endif
        }
#else
        #pragma unroll
        for (int i = 0; i < 2; ++i) {
            const int r = wrW + (i << 5);         // oc row 0..63
            u32x4 v = {};
            if ((oc0 + r) < OC && (k0 + wcW) < K)
                v = *(const u32x4*)(Wbf + (size_t)(oc0 + r) * K + k0 + wcW);
            *(u32x4*)(&Ws[bufIdx][r][wcW]) = v;
        }
#endif
    };

    auto stageP = [&](int k0, int bufIdx) {
        const int k = k0 + kcP;
        int ic = 0, khp = 0, kwp = 0;
        bool kok = (k < K);
        if (kok) {
            ic  = k >> khkwSh;
            int r = k & khkwMask;
            khp = (r >> kwSh) - pad;     // kh - pad
            kwp = (r & kwMask) - pad;    // kw - pad
        }
        #pragma unroll
        for (int i = 0; i < 32; ++i) {
            const int pr = prP + (i << 2);
            u16 v = 0;
            const int p = p0 + pr;
            if (kok && p < Mpix) {
                int b   = p >> ohowSh;
                int rem = p & ohowMask;
                int oh  = rem >> owSh;
                int ow  = rem & owMask;
                int qh  = oh * stride + khp;
                int qw  = ow * stride + kwp;
                if (dil == 1) {
                    if (qh >= 0 && qh < IH && qw >= 0 && qw < IW)
                        v = Xb[((b * IC + ic) * IH + qh) * IW + qw];
                } else {
                    if (qh >= 0 && qw >= 0 && !(qh & 1) && !(qw & 1)) {
                        int ih = qh >> 1, iw = qw >> 1;
                        if (ih < IH && iw < IW)
                            v = Xb[((b * IC + ic) * IH + ih) * IW + iw];
                    }
                }
            }
            Ps[bufIdx][pr][kcP] = v;
        }
    };

    const int nK = (K + 63) >> 6;

    // ---- prologue: stage buffer 0 ----
    stageW(0, 0);
    stageP(0, 0);
#if defined(HAVE_TDM)
    if (wv == 0) __builtin_amdgcn_s_wait_tensorcnt(0);
#endif
    __syncthreads();

    for (int kb = 0; kb < nK; ++kb) {
        const int cur = kb & 1;
        const int nxt = cur ^ 1;

        // ---- stage kb+1 into the other buffer (overlaps WMMA below) ----
        if (kb + 1 < nK) {
            stageW((kb + 1) << 6, nxt);
            stageP((kb + 1) << 6, nxt);
        }

        // ---- compute stage kb: fragments (wave32 ISA layouts), 2 K-halves ----
        const u32x4* pa0 = (const u32x4*)(&Ws[cur][wmoc + lmod][0]);
        const u32x4* pa1 = (const u32x4*)(&Ws[cur][wmoc + 16 + lmod][0]);
        const u32x4* pb0 = (const u32x4*)(&Ps[cur][wnpix + lmod][0]);
        const u32x4* pb1 = (const u32x4*)(&Ps[cur][wnpix + 16 + lmod][0]);
        #pragma unroll
        for (int kk2 = 0; kk2 < 2; ++kk2) {
            Frag fa0, fa1, fb0, fb1;
            // A 16x32 bf16: lane M=lane%16; e0..7 <- K=8*lhalf+e, e8..15 <- 16+8*lhalf+e
            const int qa = (kk2 << 2) + lhalf;
            fa0.q[0] = pa0[qa]; fa0.q[1] = pa0[qa + 2];
            fa1.q[0] = pa1[qa]; fa1.q[1] = pa1[qa + 2];
            // B 32x16 bf16: lane N=lane%16; e0..7 <- K=8*lhalf+e, e8..15 <- 16+8*lhalf+e
            const int qb = (kk2 << 2) + lhalf;
            fb0.q[0] = pb0[qb]; fb0.q[1] = pb0[qb + 2];
            fb1.q[0] = pb1[qb]; fb1.q[1] = pb1[qb + 2];

            acc00 = __builtin_amdgcn_wmma_f32_16x16x32_bf16(false, fa0.bf, false, fb0.bf,
                                                            (short)0, acc00, false, false);
            acc01 = __builtin_amdgcn_wmma_f32_16x16x32_bf16(false, fa0.bf, false, fb1.bf,
                                                            (short)0, acc01, false, false);
            acc10 = __builtin_amdgcn_wmma_f32_16x16x32_bf16(false, fa1.bf, false, fb0.bf,
                                                            (short)0, acc10, false, false);
            acc11 = __builtin_amdgcn_wmma_f32_16x16x32_bf16(false, fa1.bf, false, fb1.bf,
                                                            (short)0, acc11, false, false);
        }

#if defined(HAVE_TDM)
        if (wv == 0 && kb + 1 < nK) __builtin_amdgcn_s_wait_tensorcnt(0);
#endif
        __syncthreads();
    }

    // ---- epilogue: D layout: VGPR j -> M(oc) = j + 8*lhalf, N(pix) = lane%16 ----
    #pragma unroll
    for (int ms = 0; ms < 2; ++ms) {
        #pragma unroll
        for (int ns = 0; ns < 2; ++ns) {
            v8f acc = ms ? (ns ? acc11 : acc10) : (ns ? acc01 : acc00);
            const int pix = p0 + wnpix + (ns << 4) + lmod;
            if (pix < Mpix) {
                const int b   = pix >> ohowSh;
                const int rem = pix & ohowMask;
                #pragma unroll
                for (int j = 0; j < 8; ++j) {
                    const int oc = oc0 + wmoc + (ms << 4) + j + (lhalf << 3);
                    if (oc < OC) {
                        float v = acc[j] + bias[oc];
                        if (act == 1)      v = v > 0.0f ? v : 0.0f;
                        else if (act == 2) v = 1.0f / (1.0f + __expf(-v));
                        const size_t oi = (((size_t)(b * OC + oc)) << ohowSh) + rem;
                        if (outF) outF[oi] = v;
                        if (outB) outB[oi] = f2bf(v);
                    }
                }
            }
        }
    }
}

// ---------------------------------------------------------------------------
// VQ: per zp-row (256-dim chunk of z_e, NHWC-chunked as in reference),
// argmin_c ||c||^2 - 2 z.c over NC codes; write code row to z_q (fp32 NCHW)
// and bf16 copy for the decoder. One block per row.
// ---------------------------------------------------------------------------
__global__ __launch_bounds__(256) void vq_kernel(
    const float* __restrict__ ze, const float* __restrict__ cb,
    float* __restrict__ zq, u16* __restrict__ zqb,
    int Bn, int C, int Hs, int Ws, int D, int NC)
{
    __shared__ float z[256];
    __shared__ float sval[256];
    __shared__ int   sidx[256];

    int t   = threadIdx.x;
    int row = blockIdx.x;
    int chunks = C / D;
    int ch = row % chunks; int pr  = row / chunks;
    int ws = pr % Ws;      int pr2 = pr / Ws;
    int hs = pr2 % Hs;     int b   = pr2 / Hs;
    int cbase = ch * D;

    if (t < D) z[t] = ze[((size_t)(b * C + cbase + t) * Hs + hs) * Ws + ws];
    __syncthreads();

    float best = 3.4e38f; int bi = 0;
    for (int c = t; c < NC; c += 256) {
        const float* cr = cb + (size_t)c * D;
        float acc = 0.0f;
        for (int d = 0; d < D; ++d) {
            float cv = cr[d];
            acc += cv * (cv - 2.0f * z[d]);
        }
        if (acc < best) { best = acc; bi = c; }
    }
    sval[t] = best; sidx[t] = bi;
    __syncthreads();
    for (int s = 128; s > 0; s >>= 1) {
        if (t < s) {
            if (sval[t + s] < sval[t] ||
                (sval[t + s] == sval[t] && sidx[t + s] < sidx[t])) {
                sval[t] = sval[t + s]; sidx[t] = sidx[t + s];
            }
        }
        __syncthreads();
    }
    int bestIdx = sidx[0];
    if (t < D) {
        float cv = cb[(size_t)bestIdx * D + t];
        size_t oi = ((size_t)(b * C + cbase + t) * Hs + hs) * Ws + ws;
        zq[oi] = cv;
        if (zqb) zqb[oi] = f2bf(cv);
    }
}

// ---------------------------------------------------------------------------
// Host orchestration
// ---------------------------------------------------------------------------
extern "C" void kernel_launch(void* const* d_in, const int* in_sizes, int n_in,
                              void* d_out, int out_size, void* d_ws, size_t ws_size,
                              hipStream_t stream)
{
    (void)in_sizes; (void)n_in; (void)out_size; (void)ws_size;

    const float* x   = (const float*)d_in[0];
    const float* ew1 = (const float*)d_in[1];
    const float* eb1 = (const float*)d_in[2];
    const float* ew2 = (const float*)d_in[3];
    const float* eb2 = (const float*)d_in[4];
    const float* ew3 = (const float*)d_in[5];
    const float* eb3 = (const float*)d_in[6];
    const float* ew4 = (const float*)d_in[7];
    const float* eb4 = (const float*)d_in[8];
    const float* cb  = (const float*)d_in[9];
    const float* dw1 = (const float*)d_in[10];
    const float* db1 = (const float*)d_in[11];
    const float* tw1 = (const float*)d_in[12];
    const float* tb1 = (const float*)d_in[13];
    const float* tw2 = (const float*)d_in[14];
    const float* tb2 = (const float*)d_in[15];
    const float* tw3 = (const float*)d_in[16];
    const float* tb3 = (const float*)d_in[17];
    const float* dw2 = (const float*)d_in[18];
    const float* db2 = (const float*)d_in[19];

    float* out = (float*)d_out;
    float* xt  = out;                                 // x_tilde  (32,3,128,128)
    float* ze  = out + 1572864;                       // z_e_x    (32,512,16,16)
    float* zq  = out + 1572864 + 4194304;             // z_q_x    (32,512,16,16)

    // workspace (bf16 elements); ~372 MB total
    u16* base = (u16*)d_ws;
    size_t o = 0;
    auto alloc = [&](size_t n) { u16* p = base + o; o += n; return p; };
    u16* xb   = alloc(1572864);               // x in bf16
    u16* wp1  = alloc((size_t)256 * 48);
    u16* wp2  = alloc((size_t)256 * 4096);
    u16* wp3  = alloc((size_t)256 * 4096);
    u16* wp4  = alloc((size_t)512 * 256);
    u16* wpd1 = alloc((size_t)256 * 512);
    u16* wpt1 = alloc((size_t)256 * 4096);
    u16* wpt2 = alloc((size_t)256 * 4096);
    u16* wpt3 = alloc((size_t)256 * 4096);
    u16* wpd2 = alloc((size_t)3 * 256 + 250);
    u16* buf1 = alloc((size_t)33554432);      // (32,256,64,64)
    u16* buf2 = alloc((size_t)8388608);       // (32,256,32,32) / z_q bf16
    u16* buf3 = alloc((size_t)2097152);       // (32,256,16,16)
    u16* bufL = alloc((size_t)134217728);     // (32,256,128,128)

    // --- input + weight prep (weights -> bf16 [OC][K], flip for convT) ---
    f32_to_bf16_kernel<<<(1572864 + 255) / 256, 256, 0, stream>>>(x, xb, 1572864);
    { int n = 256 * 48;     pack_w_kernel<<<(n + 255) / 256, 256, 0, stream>>>(ew1, wp1, 3,   4, 4, 0, n); }
    { int n = 256 * 4096;   pack_w_kernel<<<(n + 255) / 256, 256, 0, stream>>>(ew2, wp2, 256, 4, 4, 0, n); }
    { int n = 256 * 4096;   pack_w_kernel<<<(n + 255) / 256, 256, 0, stream>>>(ew3, wp3, 256, 4, 4, 0, n); }
    { int n = 512 * 256;    pack_w_kernel<<<(n + 255) / 256, 256, 0, stream>>>(ew4, wp4, 256, 1, 1, 0, n); }
    { int n = 256 * 512;    pack_w_kernel<<<(n + 255) / 256, 256, 0, stream>>>(dw1, wpd1, 512, 1, 1, 0, n); }
    { int n = 256 * 4096;   pack_w_kernel<<<(n + 255) / 256, 256, 0, stream>>>(tw1, wpt1, 256, 4, 4, 1, n); }
    { int n = 256 * 4096;   pack_w_kernel<<<(n + 255) / 256, 256, 0, stream>>>(tw2, wpt2, 256, 4, 4, 1, n); }
    { int n = 256 * 4096;   pack_w_kernel<<<(n + 255) / 256, 256, 0, stream>>>(tw3, wpt3, 256, 4, 4, 1, n); }
    { int n = 3 * 256;      pack_w_kernel<<<(n + 255) / 256, 256, 0, stream>>>(dw2, wpd2, 256, 1, 1, 0, n); }

    // grid: x = OC/64 tiles, y = pixels/128 tiles
    // (owSh, ohowSh, kwSh, khkwSh, stride, pad, dil, act)
    { int M = 32 * 64 * 64; dim3 g(4, (M + 127) / 128);
      conv_wmma_kernel<<<g, 256, 0, stream>>>(xb,   wp1, eb1, nullptr, buf1,
          32, 3,   128, 128, 256, 6, 12, 2, 4, 2, 1, 1, 1); }
    { int M = 32 * 32 * 32; dim3 g(4, (M + 127) / 128);
      conv_wmma_kernel<<<g, 256, 0, stream>>>(buf1, wp2, eb2, nullptr, buf2,
          32, 256, 64,  64,  256, 5, 10, 2, 4, 2, 1, 1, 1); }
    { int M = 32 * 16 * 16; dim3 g(4, (M + 127) / 128);
      conv_wmma_kernel<<<g, 256, 0, stream>>>(buf2, wp3, eb3, nullptr, buf3,
          32, 256, 32,  32,  256, 4, 8,  2, 4, 2, 1, 1, 1); }
    { int M = 32 * 16 * 16; dim3 g(8, (M + 127) / 128);
      conv_wmma_kernel<<<g, 256, 0, stream>>>(buf3, wp4, eb4, ze, nullptr,
          32, 256, 16,  16,  512, 4, 8,  0, 0, 1, 0, 1, 0); }

    // --- vector quantisation ---
    vq_kernel<<<16384, 256, 0, stream>>>(ze, cb, zq, buf2, 32, 512, 16, 16, 256, 512);

    // --- decoder ---
    { int M = 32 * 16 * 16; dim3 g(4, (M + 127) / 128);
      conv_wmma_kernel<<<g, 256, 0, stream>>>(buf2, wpd1, db1, nullptr, buf3,
          32, 512, 16,  16,  256, 4, 8,  0, 0, 1, 0, 1, 0); }
    { int M = 32 * 32 * 32; dim3 g(4, (M + 127) / 128);
      conv_wmma_kernel<<<g, 256, 0, stream>>>(buf3, wpt1, tb1, nullptr, buf2,
          32, 256, 16,  16,  256, 5, 10, 2, 4, 1, 2, 2, 1); }
    { int M = 32 * 64 * 64; dim3 g(4, (M + 127) / 128);
      conv_wmma_kernel<<<g, 256, 0, stream>>>(buf2, wpt2, tb2, nullptr, buf1,
          32, 256, 32,  32,  256, 6, 12, 2, 4, 1, 2, 2, 1); }
    { int M = 32 * 128 * 128; dim3 g(4, (M + 127) / 128);
      conv_wmma_kernel<<<g, 256, 0, stream>>>(buf1, wpt3, tb3, nullptr, bufL,
          32, 256, 64,  64,  256, 7, 14, 2, 4, 1, 2, 2, 1); }
    { int M = 32 * 128 * 128; dim3 g(1, (M + 127) / 128);
      conv_wmma_kernel<<<g, 256, 0, stream>>>(bufL, wpd2, db2, xt, nullptr,
          32, 256, 128, 128, 3,   7, 14, 0, 0, 1, 0, 1, 2); }
}
